// InfoNCELoss_18442589569497
// MI455X (gfx1250) — matrix operand order repaired and checked by reading, not compiled, not run on hardware
//
#include <hip/hip_runtime.h>
#include <cstdint>

// ---------------------------------------------------------------------------
// InfoNCE loss, fused: normalize -> (f f^T)/T tiles via bf16 split WMMA ->
// streaming (online) logsumexp for denominator + positives -> scalar loss.
// ---------------------------------------------------------------------------

typedef __attribute__((ext_vector_type(16))) __bf16    v16bf;
typedef __attribute__((ext_vector_type(8)))  float     v8f;
typedef __attribute__((ext_vector_type(4)))  unsigned  u32x4;

union Frag16 {            // 16 bf16 = one WMMA A/B fragment per lane (8 VGPRs)
    v16bf v;
    u32x4 q[2];
};

static constexpr int   DIM    = 768;
static constexpr float INV_T  = 1.0f / 0.07f;
#define NEGINF (-__builtin_inff())

// ---------------- kernel 1: L2-normalize rows, emit bf16 hi + residual lo ---
__global__ void __launch_bounds__(256)
normalize_split_kernel(const float* __restrict__ feats,
                       uint16_t* __restrict__ FhiU, uint16_t* __restrict__ FloU)
{
    __bf16* Fhi = reinterpret_cast<__bf16*>(FhiU);
    __bf16* Flo = reinterpret_cast<__bf16*>(FloU);

    const int row = blockIdx.x;
    const int tid = threadIdx.x;
    const float* x = feats + (size_t)row * DIM;

    float ss = 0.0f;
    for (int i = tid; i < DIM; i += 256) { float t = x[i]; ss += t * t; }
    for (int m = 1; m < 32; m <<= 1) ss += __shfl_xor(ss, m, 32);

    __shared__ float red[8];
    __shared__ float s_inv;
    if ((tid & 31) == 0) red[tid >> 5] = ss;
    __syncthreads();
    if (tid == 0) {
        float t = 0.0f;
        for (int w = 0; w < 8; ++w) t += red[w];
        float n = fmaxf(sqrtf(t), 1e-12f);
        s_inv = 1.0f / n;
    }
    __syncthreads();
    const float inv = s_inv;

    for (int i = tid; i < DIM; i += 256) {
        float y   = x[i] * inv;
        __bf16 h  = (__bf16)y;               // RNE
        float  lo = y - (float)h;            // residual
        Fhi[(size_t)row * DIM + i] = h;
        Flo[(size_t)row * DIM + i] = (__bf16)lo;
    }
}

// ---------------- helpers: half-wave (16-lane) reductions, online LSE -------
__device__ __forceinline__ float halfMax(float v) {
    v = fmaxf(v, __shfl_xor(v, 1, 32));
    v = fmaxf(v, __shfl_xor(v, 2, 32));
    v = fmaxf(v, __shfl_xor(v, 4, 32));
    v = fmaxf(v, __shfl_xor(v, 8, 32));
    return v;
}
__device__ __forceinline__ float halfSum(float v) {
    v += __shfl_xor(v, 1, 32);
    v += __shfl_xor(v, 2, 32);
    v += __shfl_xor(v, 4, 32);
    v += __shfl_xor(v, 8, 32);
    return v;
}
// online logsumexp state update with one value per lane (16 lanes == one row)
__device__ __forceinline__ void onlineUpdate(float& m, float& s, float v) {
    float tmax = halfMax(v);
    float nm   = fmaxf(m, tmax);
    if (nm == NEGINF) return;                 // uniform per 16-lane half
    float p  = __expf(v - nm);                // v == -inf -> 0
    float ps = halfSum(p);
    s = s * __expf(m - nm) + ps;              // m == -inf -> scale 0
    m = nm;
}
__device__ __forceinline__ void mergePair(float& m, float& s, float m2, float s2) {
    float nm = fmaxf(m, m2);
    if (nm == NEGINF) return;
    s = s * __expf(m - nm) + s2 * __expf(m2 - nm);
    m = nm;
}

// ---------------- kernel 2: 32 rows per block, stream all column tiles ------
__global__ void __launch_bounds__(256)
infonce_kernel(const uint16_t* __restrict__ FhiU, const uint16_t* __restrict__ FloU,
               float* __restrict__ out, int N)
{
    const __bf16* Fhi = reinterpret_cast<const __bf16*>(FhiU);
    const __bf16* Flo = reinterpret_cast<const __bf16*>(FloU);

    extern __shared__ char smem[];
    __bf16* Ahi = reinterpret_cast<__bf16*>(smem);           // 32*768 bf16
    __bf16* Alo = Ahi + 32 * DIM;                            // 32*768 bf16
    float*  sMall = reinterpret_cast<float*>(Alo + 32 * DIM);// [8][32]
    float*  sSall = sMall + 8 * 32;
    float*  sMpos = sSall + 8 * 32;
    float*  sSpos = sMpos + 8 * 32;

    const int tid      = threadIdx.x;
    const int lane     = tid & 31;
    const int wid      = tid >> 5;
    const int halfsel  = lane >> 4;          // 0: lanes 0-15, 1: lanes 16-31
    const int sub      = lane & 15;
    const int rowBase0 = blockIdx.x * 32;

    // cooperative stage of the 32 A-rows (hi/lo) into LDS, 16B vectors
    {
        const u32x4* gHi = reinterpret_cast<const u32x4*>(Fhi + (size_t)rowBase0 * DIM);
        const u32x4* gLo = reinterpret_cast<const u32x4*>(Flo + (size_t)rowBase0 * DIM);
        u32x4* lHi = reinterpret_cast<u32x4*>(Ahi);
        u32x4* lLo = reinterpret_cast<u32x4*>(Alo);
        for (int i = tid; i < 32 * DIM / 8; i += 256) { lHi[i] = gHi[i]; lLo[i] = gLo[i]; }
    }
    __syncthreads();

    // running logsumexp state: 2 row-tiles x 8 C-regs, replicated in 16-lane halves
    float mAll[2][8], sAll[2][8], mPos[2][8], sPos[2][8];
#pragma unroll
    for (int rt = 0; rt < 2; ++rt)
#pragma unroll
        for (int j = 0; j < 8; ++j) {
            mAll[rt][j] = NEGINF; sAll[rt][j] = 0.0f;
            mPos[rt][j] = NEGINF; sPos[rt][j] = 0.0f;
        }

    // A fragment base pointers (LDS), per documented 16-bit A layout:
    // lane holds row M=sub, K chunks [k0 + halfsel*8, +8) and [k0+16+halfsel*8, +8)
    const __bf16* aHiB[2] = { Ahi + (0 * 16 + sub) * DIM + halfsel * 8,
                              Ahi + (1 * 16 + sub) * DIM + halfsel * 8 };
    const __bf16* aLoB[2] = { Alo + (0 * 16 + sub) * DIM + halfsel * 8,
                              Alo + (1 * 16 + sub) * DIM + halfsel * 8 };

    const int nColTiles = N >> 4;
    for (int ct = wid; ct < nColTiles; ct += 8) {     // uniform across wave
        const int colBase = ct << 4;

        v8f acc[2][3];
#pragma unroll
        for (int rt = 0; rt < 2; ++rt)
#pragma unroll
            for (int t = 0; t < 3; ++t) acc[rt][t] = (v8f){0,0,0,0,0,0,0,0};

        // B fragment: lane holds column N=sub (row of F), K chunks [k0+halfsel*8, +8) and [k0+16+halfsel*8, +8)
        const __bf16* bHiB = Fhi + (size_t)(colBase + sub) * DIM + halfsel * 8;
        const __bf16* bLoB = Flo + (size_t)(colBase + sub) * DIM + halfsel * 8;

#pragma unroll 4
        for (int k0 = 0; k0 < DIM; k0 += 32) {
            Frag16 bh, bl;
            bh.q[0] = *reinterpret_cast<const u32x4*>(bHiB + k0);
            bh.q[1] = *reinterpret_cast<const u32x4*>(bHiB + k0 + 16);
            bl.q[0] = *reinterpret_cast<const u32x4*>(bLoB + k0);
            bl.q[1] = *reinterpret_cast<const u32x4*>(bLoB + k0 + 16);
#pragma unroll
            for (int rt = 0; rt < 2; ++rt) {
                Frag16 ah, al;
                ah.q[0] = *reinterpret_cast<const u32x4*>(aHiB[rt] + k0);
                ah.q[1] = *reinterpret_cast<const u32x4*>(aHiB[rt] + k0 + 16);
                al.q[0] = *reinterpret_cast<const u32x4*>(aLoB[rt] + k0);
                al.q[1] = *reinterpret_cast<const u32x4*>(aLoB[rt] + k0 + 16);
                acc[rt][0] = __builtin_amdgcn_wmma_f32_16x16x32_bf16(
                    false, ah.v, false, bh.v, (short)0, acc[rt][0], false, false);
                acc[rt][1] = __builtin_amdgcn_wmma_f32_16x16x32_bf16(
                    false, ah.v, false, bl.v, (short)0, acc[rt][1], false, false);
                acc[rt][2] = __builtin_amdgcn_wmma_f32_16x16x32_bf16(
                    false, al.v, false, bh.v, (short)0, acc[rt][2], false, false);
            }
        }

        // epilogue: C layout -> VGPR j: lanes0-15 (M=j), lanes16-31 (M=j+8), N=sub
#pragma unroll
        for (int rt = 0; rt < 2; ++rt) {
            const int rowBase = rowBase0 + rt * 16;
            const bool nearDiag = (colBase > rowBase - 18) && (colBase < rowBase + 18);
#pragma unroll
            for (int j = 0; j < 8; ++j) {
                float v  = (acc[rt][0][j] + acc[rt][1][j] + acc[rt][2][j]) * INV_T;
                int   gR = rowBase + j + (halfsel << 3);
                int   gC = colBase + sub;
                float vd = (gR == gC) ? NEGINF : v;          // diagonal mask
                onlineUpdate(mAll[rt][j], sAll[rt][j], vd);
                if (nearDiag) {                              // positives live here
                    bool  pos = ((gR / 3) == (gC / 3)) && (gR != gC);
                    float vp  = pos ? v : NEGINF;
                    onlineUpdate(mPos[rt][j], sPos[rt][j], vp);
                }
            }
        }
    }

    // cross-wave merge: each wave publishes per-row (m,s) pairs
    if (sub == 0) {
#pragma unroll
        for (int rt = 0; rt < 2; ++rt)
#pragma unroll
            for (int j = 0; j < 8; ++j) {
                int row = rt * 16 + j + (halfsel << 3);
                sMall[wid * 32 + row] = mAll[rt][j];
                sSall[wid * 32 + row] = sAll[rt][j];
                sMpos[wid * 32 + row] = mPos[rt][j];
                sSpos[wid * 32 + row] = sPos[rt][j];
            }
    }
    __syncthreads();

    if (tid < 32) {
        const int row = tid;
        float M = NEGINF, S = 0.0f, Mp = NEGINF, Sp = 0.0f;
        for (int w = 0; w < 8; ++w) {
            mergePair(M,  S,  sMall[w * 32 + row], sSall[w * 32 + row]);
            mergePair(Mp, Sp, sMpos[w * 32 + row], sSpos[w * 32 + row]);
        }
        float loss = (M + logf(S)) - (Mp + logf(Sp));        // den - num
        for (int m = 1; m < 32; m <<= 1) loss += __shfl_xor(loss, m, 32);
        if (tid == 0) atomicAdd(out, loss / (float)N);
    }
}

// ---------------------------------------------------------------------------
extern "C" void kernel_launch(void* const* d_in, const int* in_sizes, int n_in,
                              void* d_out, int out_size, void* d_ws, size_t ws_size,
                              hipStream_t stream)
{
    const float* feats = (const float*)d_in[0];
    const int N = in_sizes[0] / DIM;                 // 6144
    uint16_t* Fhi = (uint16_t*)d_ws;                 // N*DIM bf16
    uint16_t* Flo = Fhi + (size_t)N * DIM;           // N*DIM bf16

    hipMemsetAsync(d_out, 0, sizeof(float), stream);
    normalize_split_kernel<<<N, 256, 0, stream>>>(feats, Fhi, Flo);

    const size_t shmem = (size_t)32 * DIM * 2 * 2    // Ahi + Alo (bf16)
                       + (size_t)4 * 8 * 32 * sizeof(float);
    infonce_kernel<<<N / 32, 256, shmem, stream>>>(Fhi, Flo, (float*)d_out, N);
}
